// MultiheadSelfAttention_22711787062063
// MI455X (gfx1250) — matrix operand order repaired and checked by reading, not compiled, not run on hardware
//
#include <hip/hip_runtime.h>
#include <hip/hip_fp16.h>

#define BB 4
#define SS 2048
#define HID 1024
#define NH 16
#define DD 64

typedef _Float16 v16h __attribute__((ext_vector_type(16)));
typedef float    v8f  __attribute__((ext_vector_type(8)));

// Pack two floats into packed f16 pair (low = a, high = b): v_cvt_pk_rtz_f16_f32.
__device__ __forceinline__ unsigned int pack_f16(float a, float b) {
#if __has_builtin(__builtin_amdgcn_cvt_pkrtz)
  auto t = __builtin_amdgcn_cvt_pkrtz(a, b);
  unsigned int r;
  __builtin_memcpy(&r, &t, 4);
  return r;
#else
  union { _Float16 h[2]; unsigned int u; } x;
  x.h[0] = (_Float16)a; x.h[1] = (_Float16)b;
  return x.u;
#endif
}

__device__ __forceinline__ float fast_exp2(float x) {
#if __has_builtin(__builtin_amdgcn_exp2f)
  return __builtin_amdgcn_exp2f(x);              // v_exp_f32 (base-2)
#else
  return __expf(x * 0.6931471805599453f);
#endif
}

__device__ __forceinline__ v8f wmma_f16(v16h a, v16h b, v8f c) {
  return __builtin_amdgcn_wmma_f32_16x16x32_f16(false, a, false, b, (short)0, c, false, false);
}

// A-fragment (16x32 f16): lane = m + 16*(k/16); halves = k%16
__device__ __forceinline__ v16h load_a_f16(const unsigned short* p, int ld) {
  int l = threadIdx.x & 31, m = l & 15, hi = l >> 4;
  const unsigned short* r0 = p + (size_t)m * ld + 16 * hi;
  union { v16h v; uint4 q[2]; } f;
  f.q[0] = *(const uint4*)r0;
  f.q[1] = *(const uint4*)(r0 + 8);
  return f.v;
}

// B-fragment (32x16 f16) from N-major [n][k]: lane = n + 16*(k/16); halves = k%16
__device__ __forceinline__ v16h load_b_f16(const unsigned short* p, int ld) {
  int l = threadIdx.x & 31, n = l & 15, hi = l >> 4;
  const unsigned short* r0 = p + (size_t)n * ld + 16 * hi;     // 16 contiguous k
  union { v16h v; uint4 q[2]; } f;
  f.q[0] = *(const uint4*)r0;
  f.q[1] = *(const uint4*)(r0 + 8);
  return f.v;
}

// ---------------- prep kernels (one-shot conversions) -----------------
__global__ __launch_bounds__(256) void prep_x(const float* __restrict__ x,
                                              unsigned short* __restrict__ X16) {
  int idx = blockIdx.x * 256 + threadIdx.x;                    // B*S*HID/4 threads
  float4 f = ((const float4*)x)[idx];
  uint2 w; w.x = pack_f16(f.x, f.y); w.y = pack_f16(f.z, f.w);
  ((uint2*)X16)[idx] = w;
}

__global__ __launch_bounds__(256) void prep_whead(
    const float* __restrict__ Wq, const float* __restrict__ Wk,
    const float* __restrict__ Wv,
    unsigned short* __restrict__ WQT, unsigned short* __restrict__ WKT,
    unsigned short* __restrict__ WVT) {
  int idx = blockIdx.x * 256 + threadIdx.x;                    // 3*NH*DD*DD/2 threads
  const int HALF = NH * DD * DD / 2;
  int which = idx / HALF, r = idx % HALF;
  int h = r / (DD * DD / 2), t = r % (DD * DD / 2);
  int e = t / (DD / 2), d2 = t % (DD / 2);
  const float* W = (which == 0) ? Wq : (which == 1) ? Wk : Wv;
  float v0 = W[(size_t)h * DD * DD + (2 * d2) * DD + e];
  float v1 = W[(size_t)h * DD * DD + (2 * d2 + 1) * DD + e];
  if (which == 0) { v0 *= 0.18033688f; v1 *= 0.18033688f; }    // (1/8)*log2(e)
  unsigned short* O = (which == 0) ? WQT : (which == 1) ? WKT : WVT;
  ((unsigned int*)O)[(size_t)h * DD * DD / 2 + e * (DD / 2) + d2] = pack_f16(v0, v1);
}

__global__ __launch_bounds__(256) void prep_wo(const float* __restrict__ Wo,
                                               unsigned short* __restrict__ WOT) {
  int idx = blockIdx.x * 256 + threadIdx.x;                    // HID*HID/2 threads
  int o = idx / (HID / 2), i2 = idx % (HID / 2);
  float v0 = Wo[(size_t)(2 * i2) * HID + o];
  float v1 = Wo[(size_t)(2 * i2 + 1) * HID + o];
  ((unsigned int*)WOT)[(size_t)o * (HID / 2) + i2] = pack_f16(v0, v1);
}

__global__ __launch_bounds__(256) void prep_mask(const unsigned char* __restrict__ mask,
                                                 unsigned int* __restrict__ PM) {
  int idx = blockIdx.x * 256 + threadIdx.x;                    // B*S*(S/32) threads
  const unsigned char* src = mask + (size_t)idx * 32;
  unsigned int bits = 0;
#pragma unroll
  for (int i = 0; i < 32; ++i) bits |= (src[i] ? 1u : 0u) << i;
  PM[idx] = bits;
}

// ---------------- kernel 1: Q/K/V projection (pure loads + WMMA) -----------------
__global__ __launch_bounds__(256) void qkv_kernel(
    const unsigned short* __restrict__ X16,
    const unsigned short* __restrict__ WQT, const unsigned short* __restrict__ WKT,
    const unsigned short* __restrict__ WVT,
    unsigned short* __restrict__ Q, unsigned short* __restrict__ K,
    unsigned short* __restrict__ VT) {
  int wave = threadIdx.x >> 5, l = threadIdx.x & 31;
  int hi = l >> 4, n = l & 15;
  int m0 = blockIdx.x * 128 + wave * 16;                       // token tile
  int bh = blockIdx.y, b = bh >> 4, h = bh & 15;
  const unsigned short* xb = X16 + ((size_t)(b * SS + m0)) * HID + h * DD;
  v16h xB0 = load_b_f16(xb, HID);                              // x as B (lane = token)
  v16h xB1 = load_b_f16(xb + 32, HID);
  v16h xA0 = load_a_f16(xb, HID);                              // x as A (token-major)
  v16h xA1 = load_a_f16(xb + 32, HID);
  size_t hoff = (size_t)bh * SS * DD;
  const unsigned short* WT[2] = {WQT, WKT};
  unsigned short* OUT[2] = {Q, K};
#pragma unroll
  for (int w = 0; w < 2; ++w) {
    const unsigned short* wb = WT[w] + (size_t)h * DD * DD;
#pragma unroll
    for (int j = 0; j < 4; ++j) {                              // feature tile (m dim)
      v8f c = {};
      c = wmma_f16(load_a_f16(wb + (16 * j) * DD, DD), xB0, c);
      c = wmma_f16(load_a_f16(wb + (16 * j) * DD + 32, DD), xB1, c);
      unsigned int pk[4];
#pragma unroll
      for (int i = 0; i < 4; ++i) pk[i] = pack_f16(c[2 * i], c[2 * i + 1]);
      uint4 wv; wv.x = pk[0]; wv.y = pk[1]; wv.z = pk[2]; wv.w = pk[3];
      *(uint4*)(OUT[w] + hoff + (size_t)(m0 + n) * DD + 16 * j + 8 * hi) = wv;
    }
  }
  const unsigned short* wb = WVT + (size_t)h * DD * DD;
#pragma unroll
  for (int j = 0; j < 4; ++j) {                                // feature tile (n dim)
    v8f c = {};
    c = wmma_f16(xA0, load_b_f16(wb + (16 * j) * DD, DD), c);
    c = wmma_f16(xA1, load_b_f16(wb + (16 * j) * DD + 32, DD), c);
    unsigned int pk[4];
#pragma unroll
    for (int i = 0; i < 4; ++i) pk[i] = pack_f16(c[2 * i], c[2 * i + 1]);
    uint4 wv; wv.x = pk[0]; wv.y = pk[1]; wv.z = pk[2]; wv.w = pk[3];
    *(uint4*)(VT + (size_t)bh * DD * SS + (size_t)(16 * j + n) * SS + m0 + 8 * hi) = wv;
  }
}

// Build B-fragment of P^T (32 keys x 16 queries) from two transposed score tiles
// held as packed f16 registers. 4 shuffles, no LDS.
__device__ __forceinline__ v16h make_pv_b(const unsigned int pkLo[4],
                                          const unsigned int pkHi[4], int hi) {
  union { v16h v; unsigned int q[8]; } f;
#pragma unroll
  for (int i = 0; i < 4; ++i) {
    unsigned int send = hi ? pkLo[i] : pkHi[i];
    unsigned int recv = (unsigned int)__shfl_xor((int)send, 16);
    unsigned int own  = hi ? pkHi[i] : pkLo[i];
    f.q[i]     = hi ? recv : own;
    f.q[4 + i] = hi ? own  : recv;
  }
  return f.v;
}

// ---------------- kernel 2: flash attention, transposed scores -----------------
// Two 16-query tiles per wave: every K/V fragment load feeds 2 WMMAs.
// scores^T = K x Q^T : lane = query, VGPR = key  -> per-lane softmax.
// out^T    = V^T x P^T: lane = query, VGPR = feature -> packed stores.
// Post-softmax multiplicative mask: l sums UNMASKED exp, P zeroed where masked.
__global__ __launch_bounds__(256) void attn_kernel(
    const unsigned int* __restrict__ PM,
    const unsigned short* __restrict__ Q, const unsigned short* __restrict__ K,
    const unsigned short* __restrict__ VT, unsigned short* __restrict__ AO) {
  int wave = threadIdx.x >> 5, l = threadIdx.x & 31;
  int hi = l >> 4, n = l & 15;
  int q0 = blockIdx.x * 256 + wave * 32;                       // two 16-query tiles
  int bh = blockIdx.y, b = bh >> 4, h = bh & 15;
  size_t hoff = (size_t)bh * SS * DD;
  size_t voff = (size_t)bh * DD * SS;
  v16h bq[2][2];
#pragma unroll
  for (int qt = 0; qt < 2; ++qt) {
    bq[qt][0] = load_b_f16(Q + hoff + (size_t)(q0 + 16 * qt) * DD, DD);
    bq[qt][1] = load_b_f16(Q + hoff + (size_t)(q0 + 16 * qt) * DD + 32, DD);
  }
  const unsigned int* pmrow0 = PM + ((size_t)(b * SS) + q0 + n) * (SS / 32);
  const unsigned int* pmrow1 = pmrow0 + (size_t)16 * (SS / 32);
  float mcur[2] = {-3.0e38f, -3.0e38f}, lcur[2] = {0.f, 0.f};
  v8f acc[2][4];
#pragma unroll
  for (int qt = 0; qt < 2; ++qt)
#pragma unroll
    for (int j = 0; j < 4; ++j) acc[qt][j] = (v8f){};

  for (int t0 = 0; t0 < SS; t0 += 64) {
    if (t0 + 64 < SS) {                                        // prefetch next tiles
      __builtin_prefetch(K + hoff + (size_t)(t0 + 64) * DD, 0, 3);
      __builtin_prefetch(VT + voff + t0 + 64, 0, 3);
    }
    // ---- transposed scores: each K fragment pair feeds both query tiles ----
    v8f sc[2][4];
#pragma unroll
    for (int j = 0; j < 4; ++j) {
      const unsigned short* kp = K + hoff + (size_t)(t0 + 16 * j) * DD;
      v16h aK0 = load_a_f16(kp, DD);
      v16h aK1 = load_a_f16(kp + 32, DD);
#pragma unroll
      for (int qt = 0; qt < 2; ++qt) {
        v8f c = {};
        c = wmma_f16(aK0, bq[qt][0], c);
        c = wmma_f16(aK1, bq[qt][1], c);
        sc[qt][j] = c;
      }
    }
    // ---- per-tile online softmax ----
    float corr[2];
    v16h bp[2][2];
#pragma unroll
    for (int qt = 0; qt < 2; ++qt) {
      float v = sc[qt][0][0];
#pragma unroll
      for (int j = 0; j < 4; ++j)
#pragma unroll
        for (int r = 0; r < 8; ++r) v = fmaxf(v, sc[qt][j][r]);
      v = fmaxf(v, __shfl_xor(v, 16));
      float mnew = fmaxf(mcur[qt], v);
      corr[qt] = fast_exp2(mcur[qt] - mnew);
      mcur[qt] = mnew;
      uint2 mw = *(const uint2*)((qt ? pmrow1 : pmrow0) + (t0 >> 5));
      float rs = 0.f;
      unsigned int pk[4][4];
#pragma unroll
      for (int j = 0; j < 4; ++j) {
        unsigned int word = (j < 2) ? mw.x : mw.y;
        float pm[8];
#pragma unroll
        for (int r = 0; r < 8; ++r) {
          float p = fast_exp2(sc[qt][j][r] - mnew);
          rs += p;
          int bit = 16 * (j & 1) + 8 * hi + r;
          pm[r] = ((word >> bit) & 1u) ? 0.f : p;
        }
#pragma unroll
        for (int i = 0; i < 4; ++i) pk[j][i] = pack_f16(pm[2 * i], pm[2 * i + 1]);
      }
      rs += __shfl_xor(rs, 16);
      lcur[qt] = lcur[qt] * corr[qt] + rs;
      bp[qt][0] = make_pv_b(pk[0], pk[1], hi);                 // keys t0+0..31
      bp[qt][1] = make_pv_b(pk[2], pk[3], hi);                 // keys t0+32..63
    }
    // ---- V^T @ P^T: each V fragment pair feeds both query tiles ----
#pragma unroll
    for (int jf = 0; jf < 4; ++jf) {
      const unsigned short* vp = VT + voff + (size_t)(16 * jf) * SS + t0;
      v16h aV0 = load_a_f16(vp, SS);
      v16h aV1 = load_a_f16(vp + 32, SS);
#pragma unroll
      for (int qt = 0; qt < 2; ++qt) {
        v8f a = acc[qt][jf];
#pragma unroll
        for (int r = 0; r < 8; ++r) a[r] *= corr[qt];
        a = wmma_f16(aV0, bp[qt][0], a);
        a = wmma_f16(aV1, bp[qt][1], a);
        acc[qt][jf] = a;
      }
    }
  }
  // ---- epilogue: divide by l (per-lane scalar), packed f16 stores ----
#pragma unroll
  for (int qt = 0; qt < 2; ++qt) {
    float inv = 1.f / lcur[qt];
#pragma unroll
    for (int jf = 0; jf < 4; ++jf) {
      unsigned int pk[4];
#pragma unroll
      for (int i = 0; i < 4; ++i)
        pk[i] = pack_f16(acc[qt][jf][2 * i] * inv, acc[qt][jf][2 * i + 1] * inv);
      uint4 wv; wv.x = pk[0]; wv.y = pk[1]; wv.z = pk[2]; wv.w = pk[3];
      *(uint4*)(AO + ((size_t)(b * SS + q0 + 16 * qt + n)) * HID
                + h * DD + 16 * jf + 8 * hi) = wv;
    }
  }
}

// ---------------- kernel 3: output projection GEMM (transposed) -----------------
// Two token tiles per wave: WOT A-fragments feed 2 WMMAs each.
__global__ __launch_bounds__(256) void out_gemm(
    const unsigned short* __restrict__ AO, const unsigned short* __restrict__ WOT,
    float* __restrict__ out) {
  int wave = threadIdx.x >> 5, l = threadIdx.x & 31;
  int hi = l >> 4, n = l & 15;
  int m0 = blockIdx.x * 256 + wave * 32;                       // two token tiles (n dim)
  int n0 = blockIdx.y * 64;                                    // out-feature tile (m dim)
  v8f acc[2][4];
#pragma unroll
  for (int qt = 0; qt < 2; ++qt)
#pragma unroll
    for (int j = 0; j < 4; ++j) acc[qt][j] = (v8f){};
  for (int k = 0; k < HID; k += 32) {
    v16h bx0 = load_b_f16(AO + (size_t)m0 * HID + k, HID);
    v16h bx1 = load_b_f16(AO + (size_t)(m0 + 16) * HID + k, HID);
#pragma unroll
    for (int j = 0; j < 4; ++j) {
      v16h aW = load_a_f16(WOT + (size_t)(n0 + 16 * j) * HID + k, HID);
      acc[0][j] = wmma_f16(aW, bx0, acc[0][j]);
      acc[1][j] = wmma_f16(aW, bx1, acc[1][j]);
    }
  }
#pragma unroll
  for (int qt = 0; qt < 2; ++qt)
#pragma unroll
    for (int j = 0; j < 4; ++j) {                              // 8 contiguous f32/lane
      float* op = out + (size_t)(m0 + 16 * qt + n) * HID + n0 + 16 * j + 8 * hi;
      float4 w0; w0.x = acc[qt][j][0]; w0.y = acc[qt][j][1];
      w0.z = acc[qt][j][2]; w0.w = acc[qt][j][3];
      float4 w1; w1.x = acc[qt][j][4]; w1.y = acc[qt][j][5];
      w1.z = acc[qt][j][6]; w1.w = acc[qt][j][7];
      ((float4*)op)[0] = w0;
      ((float4*)op)[1] = w1;
    }
}

extern "C" void kernel_launch(void* const* d_in, const int* in_sizes, int n_in,
                              void* d_out, int out_size, void* d_ws, size_t ws_size,
                              hipStream_t stream) {
  const float* x          = (const float*)d_in[0];
  const unsigned char* mk = (const unsigned char*)d_in[1];  // jnp bool_ -> 1 byte/elt
  const float* Wq         = (const float*)d_in[2];
  const float* Wk         = (const float*)d_in[3];
  const float* Wv         = (const float*)d_in[4];
  const float* Wo         = (const float*)d_in[5];
  (void)in_sizes; (void)n_in; (void)out_size; (void)ws_size;

  char* ws = (char*)d_ws;
  const size_t szWH = (size_t)NH * DD * DD * 2;             // 128 KB each
  unsigned short* WQT = (unsigned short*)(ws);
  unsigned short* WKT = (unsigned short*)(ws + szWH);
  unsigned short* WVT = (unsigned short*)(ws + 2 * szWH);
  unsigned short* WOT = (unsigned short*)(ws + 3 * szWH);
  size_t off = 3 * szWH + (size_t)HID * HID * 2;
  const size_t szT = (size_t)BB * NH * SS * DD * 2;         // 16 MB each
  unsigned short* X16 = (unsigned short*)(ws + off); off += szT;
  unsigned short* Q   = (unsigned short*)(ws + off); off += szT;
  unsigned short* K   = (unsigned short*)(ws + off); off += szT;
  unsigned short* VT  = (unsigned short*)(ws + off); off += szT;
  unsigned short* AO  = (unsigned short*)(ws + off); off += szT;
  unsigned int*   PM  = (unsigned int*)(ws + off);          // 2 MB bit-packed mask

  prep_x<<<(BB * SS * HID / 4) / 256, 256, 0, stream>>>(x, X16);
  prep_whead<<<(3 * NH * DD * DD / 2) / 256, 256, 0, stream>>>(Wq, Wk, Wv, WQT, WKT, WVT);
  prep_wo<<<(HID * HID / 2) / 256, 256, 0, stream>>>(Wo, WOT);
  prep_mask<<<(BB * SS * (SS / 32)) / 256, 256, 0, stream>>>(mk, PM);
  qkv_kernel<<<dim3(SS / 128, BB * NH), 256, 0, stream>>>(X16, WQT, WKT, WVT, Q, K, VT);
  attn_kernel<<<dim3(SS / 256, BB * NH), 256, 0, stream>>>(PM, Q, K, VT, AO);
  out_gemm<<<dim3(BB * SS / 256, HID / 64), 256, 0, stream>>>(AO, WOT, (float*)d_out);
}
